// DecoderGRU_50551765074375
// MI455X (gfx1250) — matrix operand order repaired and checked by reading, not compiled, not run on hardware
//
#include <hip/hip_runtime.h>
#include <cmath>

typedef __bf16 bf16_t;
typedef bf16_t v16bf __attribute__((ext_vector_type(16)));
typedef bf16_t v8bf  __attribute__((ext_vector_type(8)));
typedef float  v8f   __attribute__((ext_vector_type(8)));

#define BATCH  512
#define HIDDEN 1024
#define SEQ    64

__device__ __forceinline__ v16bf cat8(v8bf lo, v8bf hi) {
  return __builtin_shufflevector(lo, hi, 0,1,2,3,4,5,6,7,8,9,10,11,12,13,14,15);
}

__device__ __forceinline__ v8f wmma_bf16(v16bf a, v16bf b, v8f c) {
  // D(f32 16x16) = A(bf16 16x32) * B(bf16 32x16) + C
  return __builtin_amdgcn_wmma_f32_16x16x32_bf16(
      /*neg_a=*/false, a, /*neg_b=*/false, b,
      /*c_mod=*/(short)0, c, /*reuse_a=*/false, /*reuse_b=*/false);
}

// ---------------------------------------------------------------------------
// One-time (per launch) fp32 -> bf16 conversion of both weight matrices.
// ---------------------------------------------------------------------------
__global__ void gru_cvt_w(const float* __restrict__ wih, const float* __restrict__ whh,
                          bf16_t* __restrict__ wih_b, bf16_t* __restrict__ whh_b) {
  const int n = 3 * HIDDEN * HIDDEN;
  for (int i = blockIdx.x * blockDim.x + threadIdx.x; i < n; i += gridDim.x * blockDim.x) {
    wih_b[i] = (bf16_t)wih[i];
    whh_b[i] = (bf16_t)whh[i];
  }
}

__global__ void gru_init_h(const float* __restrict__ h, bf16_t* __restrict__ hb) {
  int i = blockIdx.x * blockDim.x + threadIdx.x;
  if (i < BATCH * HIDDEN) hb[i] = (bf16_t)h[i];
}

// ---------------------------------------------------------------------------
// One GRU time-step, fully fused: 6 GEMM tiles + gates in registers.
// Block = 128 threads = 4 waves; wave w owns rows [m0, m0+32) x cols
// [blockIdx.x*64 + w*16, +16) of h_new. K = HIDDEN, stepped by 32 (bf16 WMMA).
// ---------------------------------------------------------------------------
__global__ __launch_bounds__(128)
void gru_step(const bf16_t* __restrict__ hA,    // [B,H] current hidden (bf16, A-matrix)
              const bf16_t* __restrict__ wih,   // [3H,H] bf16
              const bf16_t* __restrict__ whh,   // [3H,H] bf16
              const float*  __restrict__ bih,   // [3H]
              const float*  __restrict__ bhh,   // [3H]
              const float*  __restrict__ hprev, // fp32 prev hidden, row stride hp_stride
              long hp_stride,
              float*  __restrict__ outp,        // d_out + t*H, row stride SEQ*H
              bf16_t* __restrict__ hB)          // [B,H] next hidden (bf16)
{
  const int lane = threadIdx.x & 31;
  const int wv   = threadIdx.x >> 5;   // 0..3 : N direction within workgroup
  const int half = lane >> 4;          // 0/1  : lane half
  const int nloc = lane & 15;          // column within 16-wide N tile

  const int j0 = blockIdx.x * 64 + wv * 16;  // N base within HIDDEN
  const int m0 = blockIdx.y * 32;            // M base within BATCH

  // acc[s][g]: s = M-subtile (0/1), g = {ir, iz, in, hr, hz, hn}
  v8f zero = {};
  v8f acc[2][6];
#pragma unroll
  for (int s = 0; s < 2; ++s)
#pragma unroll
    for (int g = 0; g < 6; ++g) acc[s][g] = zero;

  // A fragment base pointers (16-bit A 16x32 layout: halves at 8-elem granularity)
  const bf16_t* pa0 = hA + (size_t)(m0 + nloc) * HIDDEN + half * 8;
  const bf16_t* pa1 = pa0 + (size_t)16 * HIDDEN;

  // B fragment base pointers: lane holds weight row (j0+nloc) of each gate
  // section; halves at 8-elem granularity (same K order as A).
  const bf16_t* pwi[3];
  const bf16_t* pwh[3];
#pragma unroll
  for (int g = 0; g < 3; ++g) {
    const size_t row = (size_t)(g * HIDDEN + j0 + nloc) * HIDDEN + half * 8;
    pwi[g] = wih + row;
    pwh[g] = whh + row;
  }

  for (int k0 = 0; k0 < HIDDEN; k0 += 32) {
    v16bf a0 = cat8(*(const v8bf*)(pa0 + k0), *(const v8bf*)(pa0 + k0 + 16));
    v16bf a1 = cat8(*(const v8bf*)(pa1 + k0), *(const v8bf*)(pa1 + k0 + 16));
#pragma unroll
    for (int g = 0; g < 3; ++g) {
      v16bf bI = cat8(*(const v8bf*)(pwi[g] + k0), *(const v8bf*)(pwi[g] + k0 + 16));
      v16bf bH = cat8(*(const v8bf*)(pwh[g] + k0), *(const v8bf*)(pwh[g] + k0 + 16));
      acc[0][g]     = wmma_bf16(a0, bI, acc[0][g]);
      acc[1][g]     = wmma_bf16(a1, bI, acc[1][g]);
      acc[0][3 + g] = wmma_bf16(a0, bH, acc[0][3 + g]);
      acc[1][3 + g] = wmma_bf16(a1, bH, acc[1][3 + g]);
    }
  }

  // ------------------- fused gate epilogue -------------------
  const int j = j0 + nloc;                  // this lane's hidden index (C/D col)
  const float b_ir = bih[j];
  const float b_iz = bih[j + HIDDEN];
  const float b_in = bih[j + 2 * HIDDEN];
  const float b_hr = bhh[j];
  const float b_hz = bhh[j + HIDDEN];
  const float b_hn = bhh[j + 2 * HIDDEN];

#pragma unroll
  for (int s = 0; s < 2; ++s) {
#pragma unroll
    for (int v = 0; v < 8; ++v) {
      const int m = m0 + s * 16 + half * 8 + v;   // C/D row for VGPR v
      const float gr = (acc[s][0][v] + b_ir) + (acc[s][3][v] + b_hr);
      const float gz = (acc[s][1][v] + b_iz) + (acc[s][4][v] + b_hz);
      const float r  = 1.0f / (1.0f + __expf(-gr));
      const float z  = 1.0f / (1.0f + __expf(-gz));
      const float nn = tanhf((acc[s][2][v] + b_in) + r * (acc[s][5][v] + b_hn));
      const float hp = hprev[(size_t)m * hp_stride + j];
      const float hn = (1.0f - z) * nn + z * hp;
      outp[(size_t)m * (SEQ * HIDDEN) + j] = hn;    // d_out[b, t, :]
      hB[(size_t)m * HIDDEN + j] = (bf16_t)hn;      // next step's A matrix
    }
  }
}

extern "C" void kernel_launch(void* const* d_in, const int* in_sizes, int n_in,
                              void* d_out, int out_size, void* d_ws, size_t ws_size,
                              hipStream_t stream) {
  (void)in_sizes; (void)n_in; (void)out_size; (void)ws_size;

  const float* hidden = (const float*)d_in[0];
  const float* wih    = (const float*)d_in[1];
  const float* whh    = (const float*)d_in[2];
  const float* bih    = (const float*)d_in[3];
  const float* bhh    = (const float*)d_in[4];
  float* out = (float*)d_out;

  // Workspace layout (bytes): wih_b | whh_b | h_bf16[0] | h_bf16[1]
  char* ws = (char*)d_ws;
  const size_t wbytes = (size_t)3 * HIDDEN * HIDDEN * sizeof(bf16_t); // 6.29 MB each
  const size_t hbytes = (size_t)BATCH * HIDDEN * sizeof(bf16_t);      // 1 MB each
  bf16_t* wih_b = (bf16_t*)(ws);
  bf16_t* whh_b = (bf16_t*)(ws + wbytes);
  bf16_t* h0    = (bf16_t*)(ws + 2 * wbytes);
  bf16_t* h1    = (bf16_t*)(ws + 2 * wbytes + hbytes);

  gru_cvt_w<<<1024, 256, 0, stream>>>(wih, whh, wih_b, whh_b);
  gru_init_h<<<(BATCH * HIDDEN) / 256, 256, 0, stream>>>(hidden, h0);

  dim3 grid(HIDDEN / 64, BATCH / 32);   // 16 x 16 = 256 workgroups per step
  for (int t = 0; t < SEQ; ++t) {
    const bf16_t* hin = (t & 1) ? h1 : h0;
    bf16_t*       hot = (t & 1) ? h0 : h1;
    const float*  hp  = (t == 0) ? hidden : (out + (size_t)(t - 1) * HIDDEN);
    const long    hps = (t == 0) ? HIDDEN : (long)SEQ * HIDDEN;
    gru_step<<<grid, 128, 0, stream>>>(hin, wih_b, whh_b, bih, bhh,
                                       hp, hps, out + (size_t)t * HIDDEN, hot);
  }
}
